// SelfAttention_70196945485962
// MI455X (gfx1250) — matrix operand order repaired and checked by reading, not compiled, not run on hardware
//
#include <hip/hip_runtime.h>
#include <hip/hip_bf16.h>

typedef __attribute__((ext_vector_type(16))) __bf16 v16bf;
typedef __attribute__((ext_vector_type(8)))  __bf16 v8bf;
typedef __attribute__((ext_vector_type(8)))  float  v8f;
typedef __attribute__((ext_vector_type(4)))  float  v4f;

#define D_MODEL 1024
#define HEAD    64
#define BATCH   4
#define SEQ     2048
#define NROWS   (BATCH * SEQ)   // 8192

// ---------------------------------------------------------------------------
// helpers
// ---------------------------------------------------------------------------
__device__ __forceinline__ v16bf cat16(v8bf lo, v8bf hi) {
    v16bf r;
#pragma unroll
    for (int i = 0; i < 8; ++i) { r[i] = lo[i]; r[i + 8] = hi[i]; }
    return r;
}

__device__ __forceinline__ v8f wmma_bf16(v16bf a, v16bf b, v8f c) {
    return __builtin_amdgcn_wmma_f32_16x16x32_bf16(
        /*neg_a=*/false, a, /*neg_b=*/false, b,
        /*c_mod=*/(short)0, c, /*reuse_a=*/false, /*reuse_b=*/false);
}

// ---------------------------------------------------------------------------
// Kernel 1: transpose + convert weights  W[k][n] (fp32) -> WT[n][k] (bf16)
// ---------------------------------------------------------------------------
__global__ void prep_weights(const float* __restrict__ Wk,
                             const float* __restrict__ Wq,
                             const float* __restrict__ Wv,
                             __bf16* __restrict__ WkT,
                             __bf16* __restrict__ WqT,
                             __bf16* __restrict__ WvT) {
    int idx = blockIdx.x * blockDim.x + threadIdx.x;   // 0 .. 3*65536-1
    int w   = idx >> 16;
    int rem = idx & 0xFFFF;
    int n   = rem >> 10;        // 0..63
    int k   = rem & 1023;       // 0..1023
    const float* W  = (w == 0) ? Wk : (w == 1) ? Wq : Wv;
    __bf16*      WT = (w == 0) ? WkT : (w == 1) ? WqT : WvT;
    WT[n * D_MODEL + k] = (__bf16)W[k * HEAD + n];
}

// ---------------------------------------------------------------------------
// Kernel 2: QKV projection.  One wave computes 16 rows x 64 cols for q,k,v.
// Per k-step: all 12 loads are issued as a batch (one clause), then one wait
// covers the 4 WMMAs -- no per-WMMA stalls, no double-buffer spills.
// q scaled by 1/sqrt(D_MODEL)=1/32 row-major; k row-major; v transposed.
// ---------------------------------------------------------------------------
__global__ void qkv_proj(const float* __restrict__ x,
                         const __bf16* __restrict__ WkT,
                         const __bf16* __restrict__ WqT,
                         const __bf16* __restrict__ WvT,
                         const float* __restrict__ bk,
                         const float* __restrict__ bq,
                         const float* __restrict__ bv,
                         __bf16* __restrict__ qb,
                         __bf16* __restrict__ kb,
                         __bf16* __restrict__ vtb) {
    const int wave = threadIdx.x >> 5;
    const int lane = threadIdx.x & 31;
    const int n16  = lane & 15;
    const int hi   = lane >> 4;
    const int tile = blockIdx.x * 4 + wave;     // 0..511
    const int row0 = tile * 16;
    const int batch = row0 >> 11;

    const float* xr = x + (size_t)(row0 + n16) * D_MODEL + hi * 8;

    for (int w = 0; w < 3; ++w) {
        const __bf16* WT   = (w == 0) ? WkT : (w == 1) ? WqT : WvT;
        const float*  bias = (w == 0) ? bk  : (w == 1) ? bq  : bv;

        const __bf16* wp = WT + (size_t)n16 * D_MODEL + hi * 8;

        v8f acc[4];
#pragma unroll
        for (int nt = 0; nt < 4; ++nt)
#pragma unroll
            for (int i = 0; i < 8; ++i) acc[nt][i] = 0.f;

        for (int kc = 0; kc < D_MODEL; kc += 32) {
            // ---- batch-issue all loads for this step ----
            const float* xn = xr + kc;
            v4f f0 = *(const v4f*)(xn);
            v4f f1 = *(const v4f*)(xn + 4);
            v4f f2 = *(const v4f*)(xn + 16);
            v4f f3 = *(const v4f*)(xn + 20);
            v8bf blo[4], bhi[4];
#pragma unroll
            for (int nt = 0; nt < 4; ++nt) {
                const __bf16* bp = wp + (size_t)nt * 16 * D_MODEL + kc;
                blo[nt] = *(const v8bf*)(bp);
                bhi[nt] = *(const v8bf*)(bp + 16);
            }
            // ---- convert A, then 4 WMMAs ----
            v16bf a;
#pragma unroll
            for (int i = 0; i < 4; ++i) {
                a[i]      = (__bf16)f0[i];
                a[4 + i]  = (__bf16)f1[i];
                a[8 + i]  = (__bf16)f2[i];
                a[12 + i] = (__bf16)f3[i];
            }
#pragma unroll
            for (int nt = 0; nt < 4; ++nt)
                acc[nt] = wmma_bf16(a, cat16(blo[nt], bhi[nt]), acc[nt]);
        }

        // ---- epilogue: bias add + store ----
#pragma unroll
        for (int nt = 0; nt < 4; ++nt) {
            const int h = nt * 16 + n16;
            const float bsv = bias[h];
#pragma unroll
            for (int r = 0; r < 8; ++r) {
                float val = acc[nt][r] + bsv;
                int rowg  = row0 + r + hi * 8;
                if (w == 0) {
                    kb[(size_t)rowg * HEAD + h] = (__bf16)val;
                } else if (w == 1) {
                    qb[(size_t)rowg * HEAD + h] = (__bf16)(val * 0.03125f); // 1/sqrt(1024)
                } else {
                    vtb[(size_t)batch * HEAD * SEQ + (size_t)h * SEQ + (rowg & (SEQ - 1))] =
                        (__bf16)val;
                }
            }
        }
    }
}

// ---------------------------------------------------------------------------
// Kernel 3: causal flash attention.  One wave (32 threads) per 16 query rows.
// Per chunk: K loads (clause) -> S WMMAs -> V loads issued early -> softmax
// VALU work hides V latency -> P transpose through LDS -> PV WMMAs.
// ---------------------------------------------------------------------------
__global__ void attn(const __bf16* __restrict__ qb,
                     const __bf16* __restrict__ kb,
                     const __bf16* __restrict__ vtb,
                     float* __restrict__ out) {
    __shared__ __attribute__((aligned(16))) __bf16 Plds[16 * 40]; // row stride 40

    const int lane = threadIdx.x & 31;
    const int n16  = lane & 15;
    const int hi   = lane >> 4;
    const int bt0  = blockIdx.x * 16;      // global row of this q tile
    const int batch = bt0 >> 11;
    const int t0    = bt0 & (SEQ - 1);     // batch-local row

    const __bf16* kbase = kb  + (size_t)batch * SEQ * HEAD + (size_t)n16 * HEAD + hi * 8;
    const __bf16* vbase = vtb + (size_t)batch * HEAD * SEQ + (size_t)n16 * SEQ + hi * 8;

    // Q as A operand for the two 32-wide d-steps
    v16bf Aq[2];
#pragma unroll
    for (int d = 0; d < 2; ++d) {
        const __bf16* qp = qb + (size_t)(bt0 + n16) * HEAD + d * 32 + hi * 8;
        v8bf lo = *(const v8bf*)qp;
        v8bf hh = *(const v8bf*)(qp + 16);
        Aq[d] = cat16(lo, hh);              // A-operand chunking (p, p+16)
    }

    v8f O[4];
#pragma unroll
    for (int nt = 0; nt < 4; ++nt)
#pragma unroll
        for (int i = 0; i < 8; ++i) O[nt][i] = 0.f;

    float mrow[8], lrow[8];
#pragma unroll
    for (int r = 0; r < 8; ++r) { mrow[r] = -INFINITY; lrow[r] = 0.f; }

    const int nchunk = (t0 + 47) >> 5;     // covers keys up to diagonal
    for (int c = 0; c < nchunk; ++c) {
        const int s0 = c * 32;

        // ---- K B-operand loads for this chunk (batched clause) ----
        v8bf klo[2][2], khi[2][2];          // [st][d]
#pragma unroll
        for (int st = 0; st < 2; ++st)
#pragma unroll
            for (int d = 0; d < 2; ++d) {
                const __bf16* kp = kbase + (size_t)(s0 + st * 16) * HEAD + d * 32;
                klo[st][d] = *(const v8bf*)kp;
                khi[st][d] = *(const v8bf*)(kp + 16);
            }

        // ---- S = Q K^T for this 16x32 chunk (two 16-col subtiles) ----
        v8f S[2];
#pragma unroll
        for (int st = 0; st < 2; ++st) {
#pragma unroll
            for (int i = 0; i < 8; ++i) S[st][i] = 0.f;
#pragma unroll
            for (int d = 0; d < 2; ++d)
                S[st] = wmma_bf16(Aq[d], cat16(klo[st][d], khi[st][d]), S[st]);
        }

        // ---- issue V B-operand loads now; softmax below hides latency ----
        v8bf vlo[4], vhi[4];
#pragma unroll
        for (int nt = 0; nt < 4; ++nt) {
            const __bf16* vp = vbase + (size_t)(nt * 16) * SEQ + s0;
            vlo[nt] = *(const v8bf*)vp;
            vhi[nt] = *(const v8bf*)(vp + 16);
        }

        // ---- causal mask (only for partial chunks at the diagonal) ----
        if (s0 + 31 > t0) {
#pragma unroll
            for (int st = 0; st < 2; ++st) {
                const int s = s0 + st * 16 + n16;
#pragma unroll
                for (int r = 0; r < 8; ++r) {
                    const int t = t0 + r + hi * 8;
                    if (s > t) S[st][r] = -INFINITY;
                }
            }
        }

        // ---- online softmax: row max across the 32 keys ----
        float mnew[8], alpha[8];
#pragma unroll
        for (int r = 0; r < 8; ++r) {
            float v = fmaxf(S[0][r], S[1][r]);
            v = fmaxf(v, __shfl_xor(v, 1, 32));
            v = fmaxf(v, __shfl_xor(v, 2, 32));
            v = fmaxf(v, __shfl_xor(v, 4, 32));
            v = fmaxf(v, __shfl_xor(v, 8, 32));
            float mn = fmaxf(mrow[r], v);
            mnew[r]  = mn;
            alpha[r] = __expf(mrow[r] - mn);
            mrow[r]  = mn;
        }

        // ---- P = exp(S - m); stash to LDS (transpose to A layout) ----
#pragma unroll
        for (int st = 0; st < 2; ++st) {
#pragma unroll
            for (int r = 0; r < 8; ++r) {
                float p = __expf(S[st][r] - mnew[r]);
                S[st][r] = p;
                Plds[(r + hi * 8) * 40 + st * 16 + n16] = (__bf16)p;
            }
        }

        // ---- row sum of P ----
#pragma unroll
        for (int r = 0; r < 8; ++r) {
            float s = S[0][r] + S[1][r];
            s += __shfl_xor(s, 1, 32);
            s += __shfl_xor(s, 2, 32);
            s += __shfl_xor(s, 4, 32);
            s += __shfl_xor(s, 8, 32);
            lrow[r] = lrow[r] * alpha[r] + s;
        }

        // ---- rescale O by alpha ----
#pragma unroll
        for (int nt = 0; nt < 4; ++nt)
#pragma unroll
            for (int r = 0; r < 8; ++r) O[nt][r] *= alpha[r];

        // wait for LDS stores (CDNA5 split counter), then reload P transposed
        asm volatile("s_wait_dscnt 0" ::: "memory");
        const __bf16* pp = &Plds[n16 * 40 + hi * 8];
        v8bf plo = *(const v8bf*)pp;
        v8bf phh = *(const v8bf*)(pp + 16);
        v16bf Pa = cat16(plo, phh);         // A-operand chunking (p, p+16)

        // ---- O += P V  (V operands landed during softmax) ----
#pragma unroll
        for (int nt = 0; nt < 4; ++nt)
            O[nt] = wmma_bf16(Pa, cat16(vlo[nt], vhi[nt]), O[nt]);
    }

    // ---- epilogue: O / l -> fp32 output ----
#pragma unroll
    for (int nt = 0; nt < 4; ++nt) {
#pragma unroll
        for (int r = 0; r < 8; ++r) {
            float inv = 1.0f / lrow[r];
            out[(size_t)(bt0 + r + hi * 8) * HEAD + nt * 16 + n16] = O[nt][r] * inv;
        }
    }
}

// ---------------------------------------------------------------------------
// launch
// ---------------------------------------------------------------------------
extern "C" void kernel_launch(void* const* d_in, const int* in_sizes, int n_in,
                              void* d_out, int out_size, void* d_ws, size_t ws_size,
                              hipStream_t stream) {
    const float* x  = (const float*)d_in[0];
    const float* Wk = (const float*)d_in[1];
    const float* Wq = (const float*)d_in[2];
    const float* Wv = (const float*)d_in[3];
    const float* bk = (const float*)d_in[4];
    const float* bq = (const float*)d_in[5];
    const float* bv = (const float*)d_in[6];
    float* out = (float*)d_out;

    char* ws = (char*)d_ws;
    __bf16* qb  = (__bf16*)(ws);                                   // 1 MB
    __bf16* kb  = (__bf16*)(ws + (1u << 20));                      // 1 MB
    __bf16* vtb = (__bf16*)(ws + (2u << 20));                      // 1 MB
    __bf16* WkT = (__bf16*)(ws + (3u << 20));                      // 128 KB
    __bf16* WqT = (__bf16*)(ws + (3u << 20) + (128u << 10));       // 128 KB
    __bf16* WvT = (__bf16*)(ws + (3u << 20) + (256u << 10));       // 128 KB

    // 3 * 64 * 1024 elements / 256 threads = 768 blocks
    prep_weights<<<768, 256, 0, stream>>>(Wk, Wq, Wv, WkT, WqT, WvT);

    // 512 row-tiles of 16, 4 waves per block
    qkv_proj<<<128, 128, 0, stream>>>(x, WkT, WqT, WvT, bk, bq, bv, qb, kb, vtb);

    // one wave per 16 query rows: 8192/16 = 512 blocks
    attn<<<512, 32, 0, stream>>>(qb, kb, vtb, out);
}
